// SubmanifoldUNet_57200374448353
// MI455X (gfx1250) — matrix operand order repaired and checked by reading, not compiled, not run on hardware
//
#include <hip/hip_runtime.h>

typedef __bf16 bf16;
typedef __attribute__((ext_vector_type(16))) __bf16 v16bf;
typedef __attribute__((ext_vector_type(8)))  __bf16 v8bf;
typedef __attribute__((ext_vector_type(8)))  float  v8f;

#define TPB 256

// ---------------------------------------------------------------------------
// Scatter-mean of point features into voxels
// ---------------------------------------------------------------------------
__global__ __launch_bounds__(TPB) void k_scat_add(const float* __restrict__ feats,
    const int* __restrict__ p2v, float* __restrict__ vsum, float* __restrict__ vcnt, int P) {
  int p = blockIdx.x * blockDim.x + threadIdx.x;
  if (p >= P) return;
  int v = p2v[p];
#pragma unroll
  for (int c = 0; c < 4; ++c) atomicAdd(&vsum[v * 4 + c], feats[p * 4 + c]);
  atomicAdd(&vcnt[v], 1.0f);
}

__global__ __launch_bounds__(TPB) void k_scat_fin(const float* __restrict__ vsum,
    const float* __restrict__ vcnt, float* __restrict__ x, int N0) {
  int i = blockIdx.x * blockDim.x + threadIdx.x;
  if (i >= N0 * 4) return;
  x[i] = vsum[i] / fmaxf(vcnt[i >> 2], 1.0f);
}

// ---------------------------------------------------------------------------
// Stem conv: Cin=4 -> Cout=32 (FLOP-trivial, scalar VALU)
// ---------------------------------------------------------------------------
__global__ __launch_bounds__(TPB) void k_conv0(const float* __restrict__ x,
    const float* __restrict__ W, const int* __restrict__ nbr, float* __restrict__ y, int N) {
  int idx = blockIdx.x * blockDim.x + threadIdx.x;
  if (idx >= N * 32) return;
  int co = idx & 31, v = idx >> 5;
  float acc = 0.f;
  for (int tap = 0; tap < 27; ++tap) {
    int g = nbr[v * 27 + tap];
    if (g < 0) continue;
    const float* xg = x + g * 4;
    const float* wt = W + tap * 4 * 32 + co;
    acc += xg[0] * wt[0] + xg[1] * wt[32] + xg[2] * wt[64] + xg[3] * wt[96];
  }
  y[idx] = acc;
}

// ---------------------------------------------------------------------------
// BatchNorm statistics: LDS-reduced per-channel sum / sumsq
// ---------------------------------------------------------------------------
__global__ __launch_bounds__(TPB) void k_bn_stats(const float* __restrict__ X,
    float* __restrict__ stats, long total, int C) {
  extern __shared__ float sh[];
  for (int i = threadIdx.x; i < 2 * C; i += blockDim.x) sh[i] = 0.f;
  __syncthreads();
  long stride = (long)gridDim.x * blockDim.x;
  for (long i = (long)blockIdx.x * blockDim.x + threadIdx.x; i < total; i += stride) {
    float v = X[i];
    int c = (int)(i % C);
    atomicAdd(&sh[c], v);
    atomicAdd(&sh[C + c], v * v);
  }
  __syncthreads();
  for (int i = threadIdx.x; i < 2 * C; i += blockDim.x) atomicAdd(&stats[i], sh[i]);
}

__global__ __launch_bounds__(TPB) void k_bn_fin(const float* __restrict__ stats,
    const float* __restrict__ g, const float* __restrict__ b, float* __restrict__ ss,
    float invN, int C) {
  int c = blockIdx.x * blockDim.x + threadIdx.x;
  if (c >= C) return;
  float mu = stats[c] * invN;
  float var = stats[C + c] * invN - mu * mu;
  float sc = g[c] * rsqrtf(var + 1e-4f);
  ss[c] = sc;
  ss[C + c] = b[c] - mu * sc;
}

__global__ __launch_bounds__(TPB) void k_bnrelu_f32(const float* __restrict__ X,
    const float* __restrict__ ss, float* __restrict__ Y, long total, int C) {
  long i = (long)blockIdx.x * blockDim.x + threadIdx.x;
  if (i >= total) return;
  int c = (int)(i % C);
  Y[i] = fmaxf(X[i] * ss[c] + ss[C + c], 0.0f);
}

__global__ __launch_bounds__(TPB) void k_bnrelu_bf16(const float* __restrict__ X,
    const float* __restrict__ ss, bf16* __restrict__ Y, long total, int C) {
  long i = (long)blockIdx.x * blockDim.x + threadIdx.x;
  if (i >= total) return;
  int c = (int)(i % C);
  Y[i] = (bf16)fmaxf(X[i] * ss[c] + ss[C + c], 0.0f);
}

// ---------------------------------------------------------------------------
// Pack fp32 weights [27][Cin][Cout] into bf16 B tiles in the exact per-lane
// order the 16x16x32 bf16 WMMA B operand wants:
//   tile (tap, ko, ct): lane L holds column n = ct*16 + (L%16),
//   K values kbase + (j<8 ? 0 : 16) + (L<16 ? 0 : 8) + (j&7)  for j=0..15  (kbase = ko*32)
// ---------------------------------------------------------------------------
__global__ __launch_bounds__(TPB) void k_pack_w(const float* __restrict__ W,
    bf16* __restrict__ wp, int Cin, int Cout, int taps) {
  int KO = Cin >> 5, CT = Cout >> 4;
  long total = (long)taps * KO * CT * 512;
  long stride = (long)gridDim.x * blockDim.x;
  for (long idx = (long)blockIdx.x * blockDim.x + threadIdx.x; idx < total; idx += stride) {
    int j = (int)(idx & 15);
    int lane = (int)((idx >> 4) & 31);
    long t = idx >> 9;
    int ct = (int)(t % CT); t /= CT;
    int ko = (int)(t % KO); t /= KO;
    int tap = (int)t;
    int n = ct * 16 + (lane & 15);
    int k = ko * 32 + ((j >> 3) << 4) + ((lane >> 4) << 3) + (j & 7);
    wp[idx] = (bf16)W[((long)tap * Cin + k) * Cout + n];
  }
}

// ---------------------------------------------------------------------------
// Submanifold conv: 27-tap gather-GEMM with v_wmma_f32_16x16x32_bf16.
// One wave owns a 16-row tile x NT adjacent 16-col tiles: one gathered A tile
// feeds NT WMMAs (A-reuse cuts L2 gather traffic by NT).
// Invalid neighbors are clamped to the all-zero row at index N of xb, so the
// inner loop is branch-free (no EXEC manipulation, no register zero-fill).
// ---------------------------------------------------------------------------
template <int NT>
__global__ __launch_bounds__(TPB) void k_subm_wmma(const bf16* __restrict__ xb,
    const bf16* __restrict__ wp, const int* __restrict__ nbr,
    float* __restrict__ y, int N, int Cin, int Cout) {
  const int lane = threadIdx.x & 31;
  const int wid = threadIdx.x >> 5;
  const int colTiles = Cout >> 4;
  const int colGroups = colTiles / NT;
  const int rowTiles = (N + 15) >> 4;
  const long tile = (long)blockIdx.x * 8 + wid;
  if (tile >= (long)rowTiles * colGroups) return;   // wave-uniform: EXEC stays all-1s
  const int rt = (int)(tile / colGroups);
  const int cg = (int)(tile - (long)rt * colGroups);
  const int ct0 = cg * NT;
  const int KO = Cin >> 5;

  const int m = lane & 15;
  const int sel = lane >> 4;             // lane half selects K sub-block of A
  const int row = rt * 16 + m;
  const bool rvalid = row < N;

  v8f acc[NT];
#pragma unroll
  for (int t = 0; t < NT; ++t) acc[t] = (v8f){0.f, 0.f, 0.f, 0.f, 0.f, 0.f, 0.f, 0.f};

  for (int tap = 0; tap < 27; ++tap) {
    int g = rvalid ? nbr[row * 27 + tap] : -1;
    long grow = (g >= 0) ? (long)g : (long)N;      // row N is all zeros
    const bf16* xrow = xb + grow * Cin + sel * 8;
    for (int ko = 0; ko < KO; ++ko) {
      // A tile (16x32 bf16): lanes<16 row M=lane K {0..7,16..23}; lanes>=16 K {8..15,24..31}
      v8bf lo = *(const v8bf*)(xrow + ko * 32);
      v8bf hi = *(const v8bf*)(xrow + ko * 32 + 16);
      v16bf a;
#pragma unroll
      for (int e = 0; e < 8; ++e) { a[e] = lo[e]; a[e + 8] = hi[e]; }
      // NT B tiles (32x16 bf16 each): pre-packed, contiguous 1KB tiles
      const bf16* wb = wp + (((long)tap * KO + ko) * colTiles + ct0) * 512 + lane * 16;
#pragma unroll
      for (int t = 0; t < NT; ++t) {
        v8bf bl = *(const v8bf*)(wb + (long)t * 512);
        v8bf bh = *(const v8bf*)(wb + (long)t * 512 + 8);
        v16bf b;
#pragma unroll
        for (int e = 0; e < 8; ++e) { b[e] = bl[e]; b[e + 8] = bh[e]; }
        acc[t] = __builtin_amdgcn_wmma_f32_16x16x32_bf16(false, a, false, b,
                                                         (short)0, acc[t], false, false);
      }
    }
  }

  // D layout: VGPR j -> row rt*16 + j + sel*8, col ct*16 + lane%16
#pragma unroll
  for (int t = 0; t < NT; ++t) {
    const int col = (ct0 + t) * 16 + m;
#pragma unroll
    for (int j = 0; j < 8; ++j) {
      int r = rt * 16 + j + sel * 8;
      if (r < N) y[(long)r * Cout + col] = acc[t][j];
    }
  }
}

// ---------------------------------------------------------------------------
// Octree downsample: y[parent[v]] += x[v] @ W[oct[v]]   (atomic scatter)
// ---------------------------------------------------------------------------
__global__ __launch_bounds__(TPB) void k_down(const float* __restrict__ x,
    const float* __restrict__ W, const int* __restrict__ parent,
    const int* __restrict__ oc, float* __restrict__ y, int N, int C, int Cn) {
  long idx = (long)blockIdx.x * blockDim.x + threadIdx.x;
  if (idx >= (long)N * Cn) return;
  int co = (int)(idx % Cn);
  int v = (int)(idx / Cn);
  const float* wv = W + (long)oc[v] * C * Cn + co;
  const float* xv = x + (long)v * C;
  float a = 0.f;
  for (int ci = 0; ci < C; ++ci) a += xv[ci] * wv[(long)ci * Cn];
  atomicAdd(&y[(long)parent[v] * Cn + co], a);
}

// ---------------------------------------------------------------------------
// Octree upsample, writing directly into the right half of the concat buffer
// ---------------------------------------------------------------------------
__global__ __launch_bounds__(TPB) void k_up(const float* __restrict__ yb,
    const float* __restrict__ W, const int* __restrict__ parent,
    const int* __restrict__ oc, float* __restrict__ cat, int N, int C, int Cn) {
  long idx = (long)blockIdx.x * blockDim.x + threadIdx.x;
  if (idx >= (long)N * C) return;
  int co = (int)(idx % C);
  int v = (int)(idx / C);
  const float* wv = W + (long)oc[v] * Cn * C + co;
  const float* gv = yb + (long)parent[v] * Cn;
  float a = 0.f;
  for (int ci = 0; ci < Cn; ++ci) a += gv[ci] * wv[(long)ci * C];
  cat[(long)v * 2 * C + C + co] = a;
}

__global__ __launch_bounds__(TPB) void k_copy_left(const float* __restrict__ a,
    float* __restrict__ cat, int N, int C) {
  long idx = (long)blockIdx.x * blockDim.x + threadIdx.x;
  if (idx >= (long)N * C) return;
  int c = (int)(idx % C);
  int v = (int)(idx / C);
  cat[(long)v * 2 * C + c] = a[idx];
}

// ---------------------------------------------------------------------------
// Final bn_relu + gather back to points
// ---------------------------------------------------------------------------
__global__ __launch_bounds__(TPB) void k_out(const float* __restrict__ x0,
    const float* __restrict__ ss, const int* __restrict__ p2v,
    float* __restrict__ out, int P) {
  long idx = (long)blockIdx.x * blockDim.x + threadIdx.x;
  if (idx >= (long)P * 32) return;
  int c = (int)(idx & 31);
  int p = (int)(idx >> 5);
  int v = p2v[p];
  out[idx] = fmaxf(x0[(long)v * 32 + c] * ss[c] + ss[32 + c], 0.0f);
}

// ---------------------------------------------------------------------------
// Host orchestration
// ---------------------------------------------------------------------------
extern "C" void kernel_launch(void* const* d_in, const int* in_sizes, int n_in,
                              void* d_out, int out_size, void* d_ws, size_t ws_size,
                              hipStream_t stream) {
  (void)n_in; (void)out_size; (void)ws_size;
  static const int NP[7] = {32, 64, 96, 128, 160, 192, 224};

  // --- unpack inputs (setup_inputs() insertion order, depth-first) ---
  int cur = 0;
  const float* feats = (const float*)d_in[cur];
  const int P = in_sizes[cur] / 4; cur++;
  const float* w0 = (const float*)d_in[cur++];
  const float *pre_g[7], *pre_b[7], *pre_w[7];
  const float *down_w[6], *up_w[6], *post_w[6];
  const float *down_g[6], *down_b[6], *up_g[6], *up_b[6], *post_g[6], *post_b[6];
  for (int l = 0; l < 7; ++l) {
    pre_g[l] = (const float*)d_in[cur++];
    pre_b[l] = (const float*)d_in[cur++];
    pre_w[l] = (const float*)d_in[cur++];
    if (l < 6) {
      down_g[l] = (const float*)d_in[cur++];
      down_b[l] = (const float*)d_in[cur++];
      down_w[l] = (const float*)d_in[cur++];
      up_g[l]   = (const float*)d_in[cur++];
      up_b[l]   = (const float*)d_in[cur++];
      up_w[l]   = (const float*)d_in[cur++];
      post_g[l] = (const float*)d_in[cur++];
      post_b[l] = (const float*)d_in[cur++];
      post_w[l] = (const float*)d_in[cur++];
    }
  }
  const float* bog = (const float*)d_in[cur++];
  const float* bob = (const float*)d_in[cur++];
  const int* p2v = (const int*)d_in[cur++];
  const int* nbr[7]; int N[7];
  for (int l = 0; l < 7; ++l) { nbr[l] = (const int*)d_in[cur]; N[l] = in_sizes[cur] / 27; cur++; }
  const int* par[6]; const int* oc[6];
  for (int l = 0; l < 6; ++l) par[l] = (const int*)d_in[cur++];
  for (int l = 0; l < 6; ++l) oc[l] = (const int*)d_in[cur++];

  // --- workspace layout ---
  char* ws = (char*)d_ws;
  size_t off = 0;
  auto alloc = [&](size_t bytes) -> char* {
    char* p = ws + off; off += (bytes + 255) & ~(size_t)255; return p;
  };
  float *X[7], *A[7], *O[7];
  size_t maxNC2 = 0, maxW = 0;
  for (int l = 0; l < 7; ++l) {
    size_t nc = (size_t)N[l] * NP[l];
    X[l] = (float*)alloc(nc * 4);
    A[l] = (float*)alloc(nc * 4);
    O[l] = (float*)alloc(nc * 4);
    if (nc * 2 > maxNC2) maxNC2 = nc * 2;
    size_t w1 = (size_t)27 * NP[l] * NP[l]; if (w1 > maxW) maxW = w1;
    if (l < 6) { size_t w2 = (size_t)27 * 2 * NP[l] * NP[l]; if (w2 > maxW) maxW = w2; }
  }
  float* tF   = (float*)alloc(maxNC2 * 4);
  bf16*  tB   = (bf16*)alloc((maxNC2 + 512) * 2);   // +1 zero row (max 448 ch)
  float* cat  = (float*)alloc(maxNC2 * 4);
  bf16*  wp   = (bf16*)alloc(maxW * 2);
  float* vsum = (float*)alloc((size_t)N[0] * 4 * 4);
  float* vcnt = (float*)alloc((size_t)N[0] * 4);
  float* stats = (float*)alloc(2 * 448 * 4);
  float* ss    = (float*)alloc(2 * 448 * 4);

  auto gs = [](long total) { return (int)((total + TPB - 1) / TPB); };
  auto bn = [&](const float* Xp, const float* g, const float* b, long n, int C) {
    hipMemsetAsync(stats, 0, 2 * C * sizeof(float), stream);
    long total = n * (long)C;
    int blocks = gs(total); if (blocks > 4096) blocks = 4096;
    k_bn_stats<<<blocks, TPB, 2 * C * sizeof(float), stream>>>(Xp, stats, total, C);
    k_bn_fin<<<(C + TPB - 1) / TPB, TPB, 0, stream>>>(stats, g, b, ss, 1.0f / (float)n, C);
  };
  auto subm = [&](const float* Wp, const int* nbrp, float* yp, int n, int Cin, int Cout) {
    long tot = (long)27 * (Cin >> 5) * (Cout >> 4) * 512;
    int pblocks = gs(tot); if (pblocks > 4096) pblocks = 4096;
    k_pack_w<<<pblocks, TPB, 0, stream>>>(Wp, wp, Cin, Cout, 27);
    hipMemsetAsync(tB + (size_t)n * Cin, 0, (size_t)Cin * sizeof(bf16), stream); // zero row
    int colTiles = Cout >> 4;
    if ((colTiles & 3) == 0) {
      long tiles = (long)((n + 15) >> 4) * (colTiles >> 2);
      k_subm_wmma<4><<<(int)((tiles + 7) / 8), TPB, 0, stream>>>(tB, wp, nbrp, yp, n, Cin, Cout);
    } else {
      long tiles = (long)((n + 15) >> 4) * (colTiles >> 1);
      k_subm_wmma<2><<<(int)((tiles + 7) / 8), TPB, 0, stream>>>(tB, wp, nbrp, yp, n, Cin, Cout);
    }
  };

  // --- stem: scatter-mean + conv0 ---
  hipMemsetAsync(vsum, 0, (size_t)N[0] * 4 * 4, stream);
  hipMemsetAsync(vcnt, 0, (size_t)N[0] * 4, stream);
  k_scat_add<<<gs(P), TPB, 0, stream>>>(feats, p2v, vsum, vcnt, P);
  k_scat_fin<<<gs((long)N[0] * 4), TPB, 0, stream>>>(vsum, vcnt, tF, N[0]);
  k_conv0<<<gs((long)N[0] * 32), TPB, 0, stream>>>(tF, w0, nbr[0], X[0], N[0]);

  // --- encoder: pre-conv + down, levels 0..5 ---
  for (int l = 0; l < 6; ++l) {
    int C = NP[l], Cn = NP[l + 1];
    bn(X[l], pre_g[l], pre_b[l], N[l], C);
    k_bnrelu_bf16<<<gs((long)N[l] * C), TPB, 0, stream>>>(X[l], ss, tB, (long)N[l] * C, C);
    subm(pre_w[l], nbr[l], A[l], N[l], C, C);
    bn(A[l], down_g[l], down_b[l], N[l], C);
    k_bnrelu_f32<<<gs((long)N[l] * C), TPB, 0, stream>>>(A[l], ss, tF, (long)N[l] * C, C);
    hipMemsetAsync(X[l + 1], 0, (size_t)N[l + 1] * Cn * 4, stream);
    k_down<<<gs((long)N[l] * Cn), TPB, 0, stream>>>(tF, down_w[l], par[l], oc[l], X[l + 1], N[l], C, Cn);
  }

  // --- bottleneck: level 6 pre-conv ---
  {
    int C = NP[6];
    bn(X[6], pre_g[6], pre_b[6], N[6], C);
    k_bnrelu_bf16<<<gs((long)N[6] * C), TPB, 0, stream>>>(X[6], ss, tB, (long)N[6] * C, C);
    subm(pre_w[6], nbr[6], O[6], N[6], C, C);
  }

  // --- decoder: up + concat + post-conv, levels 5..0 ---
  for (int l = 5; l >= 0; --l) {
    int C = NP[l], Cn = NP[l + 1];
    bn(O[l + 1], up_g[l], up_b[l], N[l + 1], Cn);
    k_bnrelu_f32<<<gs((long)N[l + 1] * Cn), TPB, 0, stream>>>(O[l + 1], ss, tF, (long)N[l + 1] * Cn, Cn);
    k_up<<<gs((long)N[l] * C), TPB, 0, stream>>>(tF, up_w[l], par[l], oc[l], cat, N[l], C, Cn);
    k_copy_left<<<gs((long)N[l] * C), TPB, 0, stream>>>(A[l], cat, N[l], C);
    bn(cat, post_g[l], post_b[l], N[l], 2 * C);
    k_bnrelu_bf16<<<gs((long)N[l] * 2 * C), TPB, 0, stream>>>(cat, ss, tB, (long)N[l] * 2 * C, 2 * C);
    subm(post_w[l], nbr[l], O[l], N[l], 2 * C, C);
  }

  // --- head: bn_relu + gather back to points ---
  bn(O[0], bog, bob, N[0], 32);
  k_out<<<gs((long)P * 32), TPB, 0, stream>>>(O[0], ss, p2v, (float*)d_out, P);
}
